// AdaptiveTopKRouter_79534204387711
// MI455X (gfx1250) — matrix operand order repaired and checked by reading, not compiled, not run on hardware
//
#include <hip/hip_runtime.h>
#include <hip/hip_bf16.h>

typedef __attribute__((ext_vector_type(16))) __bf16 v16bf;
typedef __attribute__((ext_vector_type(8)))  __bf16 v8bf;
typedef __attribute__((ext_vector_type(8)))  float  v8f;

#define H_DIM 4096
#define E_DIM 64
#define TOK_PER_WG 128      // 8 waves x 16 tokens
#define KC 256              // K-chunk staged in LDS
#define B_ROW_BYTES 528     // 256 bf16 (512B) + 16B pad vs bank conflicts
#define LOGIT_STRIDE 65     // floats; conflict-free column access
// smem: max(weight stage 64*528 = 33792 B, logits 128*65*4 = 33280 B)
#define SMEM_BYTES 33792

__device__ __forceinline__ v16bf make_v16(v8bf lo, v8bf hi) {
    return __builtin_shufflevector(lo, hi, 0,1,2,3,4,5,6,7,8,9,10,11,12,13,14,15);
}

__global__ __launch_bounds__(256)
void adaptive_topk_router_kernel(const __bf16* __restrict__ hidden,
                                 const __bf16* __restrict__ weight,
                                 int* __restrict__ idx_out,
                                 __hip_bfloat16* __restrict__ w_out,
                                 int* __restrict__ k_out,
                                 int T) {
    __shared__ __align__(16) unsigned char smem[SMEM_BYTES];

    const int tid   = threadIdx.x;
    const int lane  = tid & 31;
    const int wave  = tid >> 5;           // 0..7 (wave32)
    const int wgTok = blockIdx.x * TOK_PER_WG;

    // ---------------- Phase 1: fused GEMM (logits) via WMMA ----------------
    v8f acc[4];
    #pragma unroll
    for (int nt = 0; nt < 4; ++nt) acc[nt] = (v8f){0.f,0.f,0.f,0.f,0.f,0.f,0.f,0.f};

    // Per-lane A addressing (16-bit A 16x32 layout):
    // lanes 0-15: M=lane,    K = {0..7, 16..23}
    // lanes 16-31: M=lane-16, K = {8..15, 24..31}
    const int mRow    = wgTok + wave * 16 + (lane & 15);
    const int aColOff = (lane >> 4) << 3;     // 0 or 8
    const __bf16* aRow = hidden + (size_t)mRow * H_DIM;

    for (int kc = 0; kc < H_DIM; kc += KC) {
        __syncthreads();
        // Cooperatively stage weight[0:64][kc:kc+KC] into LDS.
        // 64 rows x 32 chunks of 16B = 2048 chunks; 8 per thread.
        #pragma unroll
        for (int c = tid; c < 64 * (KC / 8); c += 256) {
            const int e  = c >> 5;        // expert row
            const int cc = c & 31;        // 16B chunk within row
            v8bf v = *(const v8bf*)(weight + (size_t)e * H_DIM + kc + cc * 8);
            *(v8bf*)(smem + e * B_ROW_BYTES + cc * 16) = v;
        }
        __syncthreads();

        // Prefetch next K-chunk of this wave's hidden rows into cache.
        if (kc + KC < H_DIM)
            __builtin_prefetch(aRow + kc + KC, 0, 0);

        for (int kl = 0; kl < KC; kl += 32) {
            const __bf16* ap = aRow + kc + kl + aColOff;
            v16bf afrag = make_v16(*(const v8bf*)ap, *(const v8bf*)(ap + 16));

            // B (32x16, 16-bit): lane holds column n=lane&15;
            // lanes 0-15 K={0..7,16..23}, lanes 16-31 K={8..15,24..31} (matches A).
            const int kb = kl + ((lane >> 4) << 3);
            #pragma unroll
            for (int nt = 0; nt < 4; ++nt) {
                const int e = nt * 16 + (lane & 15);
                const unsigned char* bp = smem + e * B_ROW_BYTES + kb * 2;
                v16bf bfrag = make_v16(*(const v8bf*)bp, *(const v8bf*)(bp + 32));
                acc[nt] = __builtin_amdgcn_wmma_f32_16x16x32_bf16(
                    false, afrag, false, bfrag, (short)0, acc[nt], false, false);
            }
        }
    }

    // ---------------- Phase 2: dump logits to LDS ----------------
    __syncthreads();           // everyone done reading staged weights
    float* lg = (float*)smem;
    {
        // C/D layout: VGPR r -> M=r (lanes 0-15) / M=r+8 (lanes 16-31); N=lane&15
        const int mOff = (lane >> 4) << 3;
        #pragma unroll
        for (int nt = 0; nt < 4; ++nt) {
            const int n = nt * 16 + (lane & 15);
            #pragma unroll
            for (int r = 0; r < 8; ++r) {
                const int t = wave * 16 + mOff + r;
                lg[t * LOGIT_STRIDE + n] = acc[nt][r];
            }
        }
    }
    __syncthreads();

    // ---------------- Phase 3: softmax / entropy / dynamic top-k ----------------
    if (tid < TOK_PER_WG) {
        const float* row = lg + tid * LOGIT_STRIDE;
        const int token = wgTok + tid;

        float m = -3.402823466e+38f;
        #pragma unroll 4
        for (int e = 0; e < E_DIM; ++e) m = fmaxf(m, row[e]);

        float S = 0.f;
        #pragma unroll 4
        for (int e = 0; e < E_DIM; ++e) S += __expf(row[e] - m);
        const float invS = 1.f / S;

        float ent = 0.f;
        float tv[4] = {-1.f, -1.f, -1.f, -1.f};
        int   ti[4] = {-1, -1, -1, -1};
        for (int e = 0; e < E_DIM; ++e) {
            const float p = __expf(row[e] - m) * invS;
            ent -= p * __logf(p + 1e-9f);
            if (p > tv[3]) {                 // strict >: first occurrence wins ties
                tv[3] = p; ti[3] = e;
                #pragma unroll
                for (int j = 3; j > 0; --j) {
                    if (tv[j] > tv[j - 1]) {
                        float fv = tv[j]; tv[j] = tv[j - 1]; tv[j - 1] = fv;
                        int   iv = ti[j]; ti[j] = ti[j - 1]; ti[j - 1] = iv;
                    }
                }
            }
        }

        int k = (ent < 0.3f) ? 1 : ((ent > 1.5f) ? 4 : 2);

        float ws = 0.f;
        #pragma unroll
        for (int j = 0; j < 4; ++j) ws += (j < k) ? tv[j] : 0.f;
        const float invW = 1.f / ws;

        #pragma unroll
        for (int j = 0; j < 4; ++j) {
            const bool act = (j < k);
            idx_out[token * 4 + j] = act ? ti[j] : -1;
            w_out[token * 4 + j]   = __float2bfloat16(act ? tv[j] * invW : 0.f);
        }
        k_out[token] = k;
    }
}

extern "C" void kernel_launch(void* const* d_in, const int* in_sizes, int n_in,
                              void* d_out, int out_size, void* d_ws, size_t ws_size,
                              hipStream_t stream) {
    const __bf16* hidden = (const __bf16*)d_in[0];   // (T, 4096) bf16
    const __bf16* weight = (const __bf16*)d_in[1];   // (64, 4096) bf16
    const int T = in_sizes[0] / H_DIM;

    // Outputs concatenated flat in return order:
    // indices (T*4 int32) | weights (T*4 bf16) | k_per_token (T int32)
    char* out = (char*)d_out;
    int*            idx_out = (int*)out;
    __hip_bfloat16* w_out   = (__hip_bfloat16*)(out + (size_t)T * 4 * sizeof(int));
    int*            k_out   = (int*)(out + (size_t)T * 4 * sizeof(int)
                                         + (size_t)T * 4 * sizeof(__hip_bfloat16));

    dim3 grid(T / TOK_PER_WG);   // 128 workgroups for T=16384
    dim3 block(256);             // 8 wave32 waves
    adaptive_topk_router_kernel<<<grid, block, 0, stream>>>(
        hidden, weight, idx_out, w_out, k_out, T);
}
